// CfCBlock_36816459661841
// MI455X (gfx1250) — matrix operand order repaired and checked by reading, not compiled, not run on hardware
//
#include <hip/hip_runtime.h>
#include <cstdint>
#include <cstddef>

// ---------------------------------------------------------------------------
// CfC block for MI455X (gfx1250, wave32, WMMA).
//   out = x + b_out + W_out * GRU( b_ih + W_ih * (b_in + W_in * LN(x)) )
// Pipeline:
//   0) convert weights fp32 -> bf16 (stay row-major [N,K])
//   1) LayerNorm  -> xn (bf16)
//   2) WMMA GEMM  u  = xn @ w_in^T  + b_in   (bf16 out)
//   3) WMMA GEMM  gx = u  @ w_ih^T  + b_ih   (bf16 out)
//   4) GRU scan: 16 WGPs, each owns 32 h-columns (96 gh columns), W_hh slice
//      LDS-resident for all 1024 steps; per step one 16x512x96 WMMA GEMM
//      (M=16 = batch), gates in fp32, h exchanged via agent-scope atomics +
//      global generation barrier.
//   5) WMMA GEMM  out = x + y @ w_out^T + b_out  (fp32 out)
//   6) h_last bf16 -> fp32 into tail of d_out
// ---------------------------------------------------------------------------

typedef unsigned short ushort_t;
typedef __attribute__((ext_vector_type(16))) __bf16 v16bf;
typedef __attribute__((ext_vector_type(8)))  float  v8f;

union Frag { v16bf v; uint4 q[2]; };

static constexpr int Bb = 16, Ss = 1024, Dd = 1024, Uu = 512;
static constexpr int Mrows = Bb * Ss;   // 16384

// workspace layout (bytes)
static constexpr size_t XN_OFF   = 0;                                  // bf16 [M,D]
static constexpr size_t U_OFF    = XN_OFF   + (size_t)Mrows * Dd * 2;  // bf16 [M,U]
static constexpr size_t GX_OFF   = U_OFF    + (size_t)Mrows * Uu * 2;  // bf16 [M,3U]
static constexpr size_t Y_OFF    = GX_OFF   + (size_t)Mrows * 3 * Uu * 2; // bf16 [M,U]
static constexpr size_t WIN_OFF  = Y_OFF    + (size_t)Mrows * Uu * 2;  // bf16 [U,D]
static constexpr size_t WIH_OFF  = WIN_OFF  + (size_t)Uu * Dd * 2;     // bf16 [3U,U]
static constexpr size_t WHH_OFF  = WIH_OFF  + (size_t)3 * Uu * Uu * 2; // bf16 [3U,U]
static constexpr size_t WOUT_OFF = WHH_OFF  + (size_t)3 * Uu * Uu * 2; // bf16 [D,U]
static constexpr size_t H_OFF    = WOUT_OFF + (size_t)Dd * Uu * 2;     // bf16 [B,U]
static constexpr size_t CNT_OFF  = H_OFF    + (size_t)Bb * Uu * 2;     // u32 barrier

__device__ __forceinline__ ushort_t f2bf(float f) {
  unsigned int u = __float_as_uint(f);
  unsigned int r = (u + 0x7FFFu + ((u >> 16) & 1u)) >> 16;   // round-nearest-even
  return (ushort_t)r;
}
__device__ __forceinline__ float bf2f(ushort_t h) {
  return __uint_as_float(((unsigned int)h) << 16);
}

// ---------------------------------------------------------------------------
__global__ void __launch_bounds__(256)
convert_f32_to_bf16(const float* __restrict__ src, ushort_t* __restrict__ dst, int n) {
  int i = blockIdx.x * blockDim.x + threadIdx.x;
  if (i < n) dst[i] = f2bf(src[i]);
}

// ---------------------------------------------------------------------------
// One 256-thread block per row of D=1024: fp32 in -> bf16 normalized out.
__global__ void __launch_bounds__(256)
layernorm_kernel(const float* __restrict__ x, const float* __restrict__ gamma,
                 const float* __restrict__ beta, ushort_t* __restrict__ xn) {
  const size_t row = blockIdx.x;
  const int tid = threadIdx.x;
  float4 v = ((const float4*)(x + row * Dd))[tid];
  float s = v.x + v.y + v.z + v.w;
  float q = v.x * v.x + v.y * v.y + v.z * v.z + v.w * v.w;
  for (int off = 16; off > 0; off >>= 1) {
    s += __shfl_xor(s, off, 32);
    q += __shfl_xor(q, off, 32);
  }
  __shared__ float ssum[8], ssq[8];
  if ((tid & 31) == 0) { ssum[tid >> 5] = s; ssq[tid >> 5] = q; }
  __syncthreads();
  float ts = 0.f, tq = 0.f;
#pragma unroll
  for (int i = 0; i < 8; ++i) { ts += ssum[i]; tq += ssq[i]; }
  float mu   = ts * (1.0f / Dd);
  float var  = tq * (1.0f / Dd) - mu * mu;
  float rstd = rsqrtf(var + 1e-5f);
  const float4 gv = ((const float4*)gamma)[tid];
  const float4 bv = ((const float4*)beta)[tid];
  float o0 = (v.x - mu) * rstd * gv.x + bv.x;
  float o1 = (v.y - mu) * rstd * gv.y + bv.y;
  float o2 = (v.z - mu) * rstd * gv.z + bv.z;
  float o3 = (v.w - mu) * rstd * gv.w + bv.w;
  uint2 pk;
  pk.x = (unsigned)f2bf(o0) | ((unsigned)f2bf(o1) << 16);
  pk.y = (unsigned)f2bf(o2) | ((unsigned)f2bf(o3) << 16);
  ((uint2*)(xn + row * Dd))[tid] = pk;
}

// ---------------------------------------------------------------------------
// WMMA GEMM: C[M,N] = A[M,K](bf16) * W[N,K](bf16)^T + bias (+ residual).
// Block = 256 threads = 8 waves (4 M x 2 N); block tile 64(M) x 128(N);
// each wave: one 16-row strip x four 16x16 N-tiles, K stepped by 32.
// Fragment loads follow the ISA 16x16x32 bf16 A/B lane layouts.
template <bool RESID>
__global__ void __launch_bounds__(256)
gemm_bf16_wmma(const ushort_t* __restrict__ A, const ushort_t* __restrict__ W,
               const float* __restrict__ bias, void* __restrict__ out,
               const float* __restrict__ residual, int M, int N, int K) {
  const int tid   = threadIdx.x;
  const int lane  = tid & 31;
  const int wid   = tid >> 5;
  const int waveM = wid & 3;
  const int waveN = wid >> 2;
  const int m0 = blockIdx.y * 64 + waveM * 16;
  const int n0 = blockIdx.x * 128 + waveN * 64;

  const int r16   = lane & 15;
  const int ahalf = (lane >> 4) ? 8 : 0;     // A: K 0..7 / 8..15 halves
  const int bhalf = (lane >> 4) ? 8 : 0;     // B: K 0..7 / 8..15 halves

  v8f acc[4] = {{}, {}, {}, {}};
  for (int k0 = 0; k0 < K; k0 += 32) {
    Frag a;
    const ushort_t* ap = A + (size_t)(m0 + r16) * K + k0 + ahalf;
    a.q[0] = *(const uint4*)ap;
    a.q[1] = *(const uint4*)(ap + 16);
#pragma unroll
    for (int j = 0; j < 4; ++j) {
      Frag b;
      const ushort_t* bp = W + (size_t)(n0 + j * 16 + r16) * K + k0 + bhalf;
      b.q[0] = *(const uint4*)bp;
      b.q[1] = *(const uint4*)(bp + 16);
      acc[j] = __builtin_amdgcn_wmma_f32_16x16x32_bf16(
          false, a.v, false, b.v, (short)0, acc[j], false, false);
    }
  }
  const int mh = (lane >> 4) ? 8 : 0;
#pragma unroll
  for (int j = 0; j < 4; ++j) {
    int n = n0 + j * 16 + r16;
    float bv = bias[n];
#pragma unroll
    for (int r = 0; r < 8; ++r) {
      int m = m0 + r + mh;
      float v = acc[j][r] + bv;
      if (RESID) {
        ((float*)out)[(size_t)m * N + n] = v + residual[(size_t)m * N + n];
      } else {
        ((ushort_t*)out)[(size_t)m * N + n] = f2bf(v);
      }
    }
  }
}

// ---------------------------------------------------------------------------
__global__ void __launch_bounds__(256)
gru_init_kernel(unsigned int* __restrict__ h_u32, unsigned int* __restrict__ cnt) {
  int i = blockIdx.x * blockDim.x + threadIdx.x;
  if (i < (Bb * Uu * 2) / 4) h_u32[i] = 0u;
  if (i == 0) cnt[0] = 0u;
}

// ---------------------------------------------------------------------------
// GRU scan. 16 blocks (one per WGP); block g owns h columns [g*32, g*32+32)
// and gh columns {r,z,n} x that slice (6 16-col WMMA tiles, waves 0..5).
// W_hh slice (96x512 bf16, padded stride 520 -> conflict-free ds_load_b128)
// lives in LDS for the whole scan. Per step: agent-scope h gather -> LDS,
// 16x512x96 WMMA, fused gates, scoped h scatter, global generation barrier.
__global__ void __launch_bounds__(256)
gru_scan_kernel(const ushort_t* __restrict__ gx,   // [M,3U] bf16 (b_ih folded in)
                const ushort_t* __restrict__ whh,  // [3U,U]  bf16
                const float*    __restrict__ bhh,  // [3U]    fp32
                ushort_t* __restrict__ y,          // [M,U]   bf16
                ushort_t* __restrict__ gH,         // [B,U]   bf16 (coherent)
                unsigned int* __restrict__ cnt) {
  constexpr int NWG = 16, COLS = 32;
  constexpr int WROW = 520;                         // padded ushort stride
  extern __shared__ char smem[];
  ushort_t* w_lds  = (ushort_t*)smem;               // [96][520]
  ushort_t* h_lds  = w_lds + 96 * WROW;             // [16][520]
  float*    gh_lds = (float*)(h_lds + 16 * WROW);   // [16][100]

  const int tid  = threadIdx.x;
  const int lane = tid & 31;
  const int wave = tid >> 5;
  const int g    = blockIdx.x;

  // Stage W_hh slice: local rows 0-31 = r gate, 32-63 = z, 64-95 = n.
  for (int idx = tid; idx < 96 * 64; idx += 256) {
    int lr = idx >> 6;
    int ck = (idx & 63) * 8;
    int gate = lr >> 5;
    int grow = gate * Uu + g * COLS + (lr & 31);
    *(uint4*)&w_lds[lr * WROW + ck] = *(const uint4*)&whh[(size_t)grow * Uu + ck];
  }
  __syncthreads();

  const int arow  = lane & 15;
  const int ahalf = (lane >> 4) ? 8 : 0;
  const int bhalf = (lane >> 4) ? 8 : 0;

  for (int t = 0; t < Ss; ++t) {
    // 1) gather full h[16][512] (bf16) with agent scope (bypass WGP$) -> LDS
#pragma unroll
    for (int i = 0; i < 8; ++i) {
      int idx = tid + i * 256;            // 2048 x 8B = 16x512 bf16
      int row = idx >> 7;
      int c64 = idx & 127;
      unsigned long long vv = __hip_atomic_load(
          (const unsigned long long*)gH + (size_t)row * 128 + c64,
          __ATOMIC_RELAXED, __HIP_MEMORY_SCOPE_AGENT);
      *(unsigned long long*)&h_lds[row * WROW + c64 * 4] = vv;
    }
    __syncthreads();

    // 2) gh(16x96) = h(16x512) x Whh_slice^T : 6 waves x 16 WMMAs
    if (wave < 6) {
      v8f acc = {};
#pragma unroll
      for (int kt = 0; kt < 16; ++kt) {
        Frag a, b;
        const ushort_t* ap = &h_lds[arow * WROW + kt * 32 + ahalf];
        a.q[0] = *(const uint4*)ap;
        a.q[1] = *(const uint4*)(ap + 16);
        const ushort_t* bp = &w_lds[(wave * 16 + arow) * WROW + kt * 32 + bhalf];
        b.q[0] = *(const uint4*)bp;
        b.q[1] = *(const uint4*)(bp + 16);
        acc = __builtin_amdgcn_wmma_f32_16x16x32_bf16(
            false, a.v, false, b.v, (short)0, acc, false, false);
      }
      const int mh = (lane >> 4) ? 8 : 0;
#pragma unroll
      for (int r = 0; r < 8; ++r)
        gh_lds[(r + mh) * 100 + wave * 16 + arow] = acc[r];
    }
    __syncthreads();

    // 3) gates + state update (fp32) for 16 batch x 32 columns
    for (int o = tid; o < Bb * COLS; o += 256) {
      int b = o >> 5;
      int c = o & (COLS - 1);
      int gc = g * COLS + c;
      size_t gxrow = ((size_t)b * Ss + t) * (3 * Uu);
      float ghr = gh_lds[b * 100 + c]          + bhh[gc];
      float ghz = gh_lds[b * 100 + 32 + c]     + bhh[Uu + gc];
      float ghn = gh_lds[b * 100 + 64 + c]     + bhh[2 * Uu + gc];
      float xr  = bf2f(gx[gxrow + gc]);
      float xz  = bf2f(gx[gxrow + Uu + gc]);
      float xnv = bf2f(gx[gxrow + 2 * Uu + gc]);
      float rg = 1.f / (1.f + __expf(-(xr + ghr)));
      float zg = 1.f / (1.f + __expf(-(xz + ghz)));
      float ng = tanhf(xnv + rg * ghn);
      float hold = bf2f(h_lds[b * WROW + gc]);
      float hnew = (1.f - zg) * ng + zg * hold;
      ushort_t hb = f2bf(hnew);
      y[((size_t)b * Ss + t) * Uu + gc] = hb;
      __hip_atomic_store(gH + (size_t)b * Uu + gc, hb,
                         __ATOMIC_RELAXED, __HIP_MEMORY_SCOPE_AGENT);
      if (t + 1 < Ss)
        __builtin_prefetch(&gx[gxrow + 3 * Uu + gc], 0, 1);   // global_prefetch
    }
    __threadfence();
    __syncthreads();

    // 4) 16-workgroup generation barrier (monotonic counter)
    if (tid == 0) {
      __hip_atomic_fetch_add(cnt, 1u, __ATOMIC_ACQ_REL, __HIP_MEMORY_SCOPE_AGENT);
      unsigned int target = (unsigned)NWG * (unsigned)(t + 1);
      while (__hip_atomic_load(cnt, __ATOMIC_ACQUIRE, __HIP_MEMORY_SCOPE_AGENT) < target)
        __builtin_amdgcn_s_sleep(2);
    }
    __syncthreads();
  }
}

// ---------------------------------------------------------------------------
__global__ void __launch_bounds__(256)
hlast_kernel(const ushort_t* __restrict__ h, float* __restrict__ out) {
  int i = blockIdx.x * blockDim.x + threadIdx.x;
  if (i < Bb * Uu) out[i] = bf2f(h[i]);
}

// ---------------------------------------------------------------------------
extern "C" void kernel_launch(void* const* d_in, const int* in_sizes, int n_in,
                              void* d_out, int out_size, void* d_ws, size_t ws_size,
                              hipStream_t stream) {
  const float* x     = (const float*)d_in[0];
  const float* gam   = (const float*)d_in[1];
  const float* bet   = (const float*)d_in[2];
  const float* w_in  = (const float*)d_in[3];
  const float* b_in  = (const float*)d_in[4];
  const float* w_ih  = (const float*)d_in[5];
  const float* b_ih  = (const float*)d_in[6];
  const float* w_hh  = (const float*)d_in[7];
  const float* b_hh  = (const float*)d_in[8];
  const float* w_out = (const float*)d_in[9];
  const float* b_out = (const float*)d_in[10];

  char* ws = (char*)d_ws;
  ushort_t* xn   = (ushort_t*)(ws + XN_OFF);
  ushort_t* u    = (ushort_t*)(ws + U_OFF);
  ushort_t* gx   = (ushort_t*)(ws + GX_OFF);
  ushort_t* y    = (ushort_t*)(ws + Y_OFF);
  ushort_t* win  = (ushort_t*)(ws + WIN_OFF);
  ushort_t* wih  = (ushort_t*)(ws + WIH_OFF);
  ushort_t* whh  = (ushort_t*)(ws + WHH_OFF);
  ushort_t* wout = (ushort_t*)(ws + WOUT_OFF);
  ushort_t* gH   = (ushort_t*)(ws + H_OFF);
  unsigned int* cnt = (unsigned int*)(ws + CNT_OFF);

  float* outp  = (float*)d_out;
  float* hlast = outp + (size_t)Bb * Ss * Dd;

  // 0) weights fp32 -> bf16 (row-major [N,K] preserved)
  convert_f32_to_bf16<<<(Uu * Dd + 255) / 256, 256, 0, stream>>>(w_in, win, Uu * Dd);
  convert_f32_to_bf16<<<(3 * Uu * Uu + 255) / 256, 256, 0, stream>>>(w_ih, wih, 3 * Uu * Uu);
  convert_f32_to_bf16<<<(3 * Uu * Uu + 255) / 256, 256, 0, stream>>>(w_hh, whh, 3 * Uu * Uu);
  convert_f32_to_bf16<<<(Dd * Uu + 255) / 256, 256, 0, stream>>>(w_out, wout, Dd * Uu);

  // 1) LayerNorm
  layernorm_kernel<<<Mrows, 256, 0, stream>>>(x, gam, bet, xn);

  // 2) u = xn @ w_in^T + b_in   (M=16384, N=512, K=1024)
  gemm_bf16_wmma<false><<<dim3(Uu / 128, Mrows / 64), 256, 0, stream>>>(
      xn, win, b_in, (void*)u, nullptr, Mrows, Uu, Dd);

  // 3) gx = u @ w_ih^T + b_ih   (M=16384, N=1536, K=512)
  gemm_bf16_wmma<false><<<dim3(3 * Uu / 128, Mrows / 64), 256, 0, stream>>>(
      u, wih, b_ih, (void*)gx, nullptr, Mrows, 3 * Uu, Uu);

  // 4) GRU scan (h0 = 0; barrier counter reset every launch)
  gru_init_kernel<<<16, 256, 0, stream>>>((unsigned int*)gH, cnt);
  constexpr size_t GRU_LDS = (size_t)96 * 520 * 2 + 16 * 520 * 2 + 16 * 100 * 4; // 122880
  gru_scan_kernel<<<16, 256, GRU_LDS, stream>>>(gx, whh, b_hh, y, gH, cnt);

  // 5) out = x + y @ w_out^T + b_out   (M=16384, N=1024, K=512, fp32 out)
  gemm_bf16_wmma<true><<<dim3(Dd / 128, Mrows / 64), 256, 0, stream>>>(
      y, wout, b_out, (void*)outp, x, Mrows, Dd, Uu);

  // 6) h_last
  hlast_kernel<<<(Bb * Uu + 255) / 256, 256, 0, stream>>>(gH, hlast);
}
